// ComplexApproximateBNN_82214263980129
// MI455X (gfx1250) — matrix operand order repaired and checked by reading, not compiled, not run on hardware
//
#include <hip/hip_runtime.h>
#include <hip/hip_bf16.h>
#include <math.h>

#define B_    64
#define T_    128
#define IN_   512
#define X_    2048
#define OUT_  256
#define NWG_  32
#define NTHR_ 256

typedef __bf16 bf16;
typedef bf16  v16bf __attribute__((ext_vector_type(16)));
typedef bf16  v8bf  __attribute__((ext_vector_type(8)));
typedef float v8f   __attribute__((ext_vector_type(8)));

struct Ws {
  int*   cnt;      // grid barrier counter
  float* ss_li;    // [B] sum of squares for li (h2)
  float* ss_bp;    // [B] sum of squares for bp (h4)
  bf16*  xbf;      // [T][B][IN]
  bf16*  w_in;     // [X][IN]
  bf16*  w_h;      // [4][X][2X]
  bf16*  w_out;    // [OUT][X]
  bf16*  h0; bf16* h1; bf16* h2; bf16* h3; bf16* h4;  // [B][X]
  bf16*  bp; bf16* li;                                 // [B][X]
};

__device__ __forceinline__ float apply_act(float h, int id) {
  switch (id) {
    case 0:  return fmaxf(h, 0.0f);                       // relu
    case 1:  return 1.0f / (1.0f + __expf(-h));           // sigmoid
    case 2:  return tanhf(h);                             // tanh
    case 3:  return h >= 0.0f ? h : 0.1f * h;             // leaky relu
    default: {                                            // selu
      const float a = 1.6732632423543772f, s = 1.0507009873554805f;
      return h > 0.0f ? s * h : s * a * (__expf(h) - 1.0f);
    }
  }
}

__device__ __forceinline__ void grid_barrier(int* cnt, int round) {
  __threadfence();            // make this block's global stores device-visible
  __syncthreads();
  if (threadIdx.x == 0) {
    __hip_atomic_fetch_add(cnt, 1, __ATOMIC_RELEASE, __HIP_MEMORY_SCOPE_AGENT);
    const int target = (int)gridDim.x * round;
    while (__hip_atomic_load(cnt, __ATOMIC_ACQUIRE, __HIP_MEMORY_SCOPE_AGENT) < target)
      __builtin_amdgcn_s_sleep(1);
  }
  __syncthreads();
}

// D[b, n] = act( sum_k cat(A0,A1)[b,k] * W[n,k] + bias[n] )
// A0: [64][K0], A1: [64][K1] (optional), W: [N][K0+K1] row-major bf16.
// Wave computes 2 M-tiles (rows mg*32 .. mg*32+31) x 1 N-tile (16 cols).
__device__ __forceinline__ void gemm_phase(
    const bf16* __restrict__ A0, int K0,
    const bf16* __restrict__ A1, int K1,
    const bf16* __restrict__ W, int N,
    const float* __restrict__ bias, const int* __restrict__ actid,
    bf16* __restrict__ Hout, float* __restrict__ yout, int t,
    float* __restrict__ sumsq) {
  const int lane   = threadIdx.x & 31;
  const int gwave  = blockIdx.x * (blockDim.x >> 5) + (threadIdx.x >> 5);
  const int nwaves = gridDim.x * (blockDim.x >> 5);
  const int K      = K0 + K1;
  const int njobs  = (N >> 4) * 2;
  const int lhalf  = lane >> 4;       // 0 or 1
  const int l15    = lane & 15;

  for (int job = gwave; job < njobs; job += nwaves) {
    const int nstrip = job >> 1;
    const int mg     = job & 1;
    const int n      = (nstrip << 4) + l15;
    // B fragment: lane holds K = k0+{0|8}..+7 and k0+{16|24}..+7 of W row n (matches A layout)
    const bf16* wrow = W + (size_t)n * K + (lhalf << 3);
    // A fragment: lane holds K = k0+{0|8}..+7 and k0+{16|24}..+7 of its row (ISA A layout)
    const int arow0 = mg * 32 + l15;
    const int aoff  = lhalf << 3;

    v8f c0 = {};
    v8f c1 = {};
    for (int k0 = 0; k0 < K; k0 += 32) {
      const bf16* Ab; int rs;
      if (k0 < K0) { Ab = A0 + k0; rs = K0; }
      else         { Ab = A1 + (k0 - K0); rs = K1; }
      const bf16* ap = Ab + (size_t)arow0 * rs + aoff;
      v8bf x0 = *(const v8bf*)(ap);
      v8bf x1 = *(const v8bf*)(ap + 16);
      const bf16* aq = ap + (size_t)16 * rs;     // second M-tile: rows +16
      v8bf x2 = *(const v8bf*)(aq);
      v8bf x3 = *(const v8bf*)(aq + 16);
      v16bf a0 = __builtin_shufflevector(x0, x1, 0,1,2,3,4,5,6,7,8,9,10,11,12,13,14,15);
      v16bf a1 = __builtin_shufflevector(x2, x3, 0,1,2,3,4,5,6,7,8,9,10,11,12,13,14,15);
      v8bf y0 = *(const v8bf*)(wrow + k0);
      v8bf y1 = *(const v8bf*)(wrow + k0 + 16);
      v16bf bfrag = __builtin_shufflevector(y0, y1, 0,1,2,3,4,5,6,7,8,9,10,11,12,13,14,15);
      c0 = __builtin_amdgcn_wmma_f32_16x16x32_bf16(false, a0, false, bfrag, (short)0, c0, false, false);
      c1 = __builtin_amdgcn_wmma_f32_16x16x32_bf16(false, a1, false, bfrag, (short)0, c1, false, false);
    }

    const float bi  = bias[n];
    const int   aid = actid[n];
    const int   rb  = mg * 32 + (lhalf << 3);  // C/D layout: lanes 0-15 -> M=r, 16-31 -> M=8+r
#pragma unroll
    for (int mt = 0; mt < 2; ++mt) {
      v8f c = mt ? c1 : c0;
#pragma unroll
      for (int r = 0; r < 8; ++r) {
        const int row = rb + mt * 16 + r;
        float v = apply_act(c[r] + bi, aid);
        if (yout) {
          yout[((size_t)row * T_ + t) * OUT_ + n] = v;   // y[b][t][o]
        } else {
          Hout[(size_t)row * N + n] = (bf16)v;
        }
        if (sumsq) {
          float vb = (float)(bf16)v;   // norm the stored (rounded) vector
          float pp = vb * vb;
          pp += __shfl_xor(pp, 1);
          pp += __shfl_xor(pp, 2);
          pp += __shfl_xor(pp, 4);
          pp += __shfl_xor(pp, 8);
          if (l15 == 0) atomicAdd(&sumsq[row], pp);
        }
      }
    }
  }
}

// dst[b][k] = bf16( src[b][k] / max(||src[b]||, 1e-12) )
__device__ __forceinline__ void normalize_copy(const bf16* __restrict__ src,
                                               const float* __restrict__ ss,
                                               bf16* __restrict__ dst) {
  const int gtid = blockIdx.x * blockDim.x + threadIdx.x;
  const int gsz  = gridDim.x * blockDim.x;
  for (int i = gtid; i < B_ * X_; i += gsz) {
    const int row = i >> 11;   // X_ = 2048
    const float scale = 1.0f / fmaxf(sqrtf(ss[row]), 1e-12f);
    dst[i] = (bf16)((float)src[i] * scale);
  }
}

__global__ void __launch_bounds__(NTHR_, 1)
bnn_persistent(Ws ws, const float* __restrict__ b_in, const float* __restrict__ b_h,
               const float* __restrict__ b_out, const int* __restrict__ act_ids,
               const int* __restrict__ out_act_ids, float* __restrict__ y) {
  const int gtid = blockIdx.x * blockDim.x + threadIdx.x;
  int round = 1;
  for (int t = 0; t < T_; ++t) {
    // G0: h0 = act0(x_t @ W_in^T + b_in)
    gemm_phase(ws.xbf + (size_t)t * B_ * IN_, IN_, nullptr, 0, ws.w_in, X_,
               b_in, act_ids, ws.h0, nullptr, 0, nullptr);
    grid_barrier(ws.cnt, round++);

    // G1: h1 = act1(cat(h0, bp) @ Wh0^T + bh0); also reset ss_bp for this step
    if (gtid < B_) ws.ss_bp[gtid] = 0.0f;
    gemm_phase(ws.h0, X_, ws.bp, X_, ws.w_h, X_,
               b_h, act_ids + X_, ws.h1, nullptr, 0, nullptr);
    grid_barrier(ws.cnt, round++);

    // G2: h2 = act2(cat(h1, li) @ Wh1^T + bh1), accumulate ss_li
    gemm_phase(ws.h1, X_, ws.li, X_, ws.w_h + (size_t)1 * X_ * 2 * X_, X_,
               b_h + X_, act_ids + 2 * X_, ws.h2, nullptr, 0, ws.ss_li);
    grid_barrier(ws.cnt, round++);

    // G3: h3 = act3(cat(h2, h1) @ Wh2^T + bh2); in parallel, li <- l2norm(h2)
    gemm_phase(ws.h2, X_, ws.h1, X_, ws.w_h + (size_t)2 * X_ * 2 * X_, X_,
               b_h + 2 * X_, act_ids + 3 * X_, ws.h3, nullptr, 0, nullptr);
    normalize_copy(ws.h2, ws.ss_li, ws.li);
    grid_barrier(ws.cnt, round++);

    // G4: h4 = act4(cat(h3, h2) @ Wh3^T + bh3), accumulate ss_bp; reset ss_li
    if (gtid < B_) ws.ss_li[gtid] = 0.0f;
    gemm_phase(ws.h3, X_, ws.h2, X_, ws.w_h + (size_t)3 * X_ * 2 * X_, X_,
               b_h + 3 * X_, act_ids + 4 * X_, ws.h4, nullptr, 0, ws.ss_bp);
    grid_barrier(ws.cnt, round++);

    // OUT: y_t = act_out(h4 @ W_out^T + b_out); in parallel, bp <- l2norm(h4)
    gemm_phase(ws.h4, X_, nullptr, 0, ws.w_out, OUT_,
               b_out, out_act_ids, nullptr, y, t, nullptr);
    normalize_copy(ws.h4, ws.ss_bp, ws.bp);
    grid_barrier(ws.cnt, round++);
  }
}

// ---- prep kernels (run every launch; graph-capture safe) ----
__global__ void k_prep_weights(const float* __restrict__ Win, const float* __restrict__ Wh,
                               const float* __restrict__ Wout, Ws ws) {
  const size_t nin = (size_t)X_ * IN_;
  const size_t nh  = (size_t)4 * X_ * 2 * X_;
  const size_t nout = (size_t)OUT_ * X_;
  const size_t total = nin + nh + nout;
  const size_t gsz = (size_t)gridDim.x * blockDim.x;
  for (size_t i = (size_t)blockIdx.x * blockDim.x + threadIdx.x; i < total; i += gsz) {
    if (i < nin)            ws.w_in[i]            = (bf16)Win[i];
    else if (i < nin + nh)  ws.w_h[i - nin]       = (bf16)Wh[i - nin];
    else                    ws.w_out[i - nin - nh] = (bf16)Wout[i - nin - nh];
  }
}

__global__ void k_prep_x(const float* __restrict__ x, Ws ws) {
  // x: [B][T][IN] f32  ->  xbf: [T][B][IN] bf16
  const size_t total = (size_t)B_ * T_ * IN_;
  const size_t gsz = (size_t)gridDim.x * blockDim.x;
  for (size_t i = (size_t)blockIdx.x * blockDim.x + threadIdx.x; i < total; i += gsz) {
    const size_t b = i / ((size_t)T_ * IN_);
    const size_t rem = i % ((size_t)T_ * IN_);
    const size_t t = rem / IN_;
    const size_t k = rem % IN_;
    ws.xbf[(t * B_ + b) * IN_ + k] = (bf16)x[i];
  }
}

__global__ void k_prep_state(Ws ws) {
  const int gtid = blockIdx.x * blockDim.x + threadIdx.x;
  const int gsz  = gridDim.x * blockDim.x;
  for (int i = gtid; i < B_ * X_; i += gsz) { ws.bp[i] = (bf16)0.0f; ws.li[i] = (bf16)0.0f; }
  if (gtid < B_) { ws.ss_li[gtid] = 0.0f; ws.ss_bp[gtid] = 0.0f; }
  if (gtid == 0) { *ws.cnt = 0; }
}

extern "C" void kernel_launch(void* const* d_in, const int* in_sizes, int n_in,
                              void* d_out, int out_size, void* d_ws, size_t ws_size,
                              hipStream_t stream) {
  const float* x     = (const float*)d_in[0];
  const float* W_in  = (const float*)d_in[1];
  const float* b_in  = (const float*)d_in[2];
  const float* W_h   = (const float*)d_in[3];
  const float* b_h   = (const float*)d_in[4];
  const float* W_out = (const float*)d_in[5];
  const float* b_out = (const float*)d_in[6];
  const int* act_ids     = (const int*)d_in[7];
  const int* out_act_ids = (const int*)d_in[8];
  float* y = (float*)d_out;

  size_t off = 0;
  auto take = [&](size_t bytes) -> void* {
    void* r = (char*)d_ws + off;
    off += (bytes + 255) & ~(size_t)255;
    return r;
  };
  Ws ws;
  ws.cnt   = (int*)  take(256);
  ws.ss_li = (float*)take(B_ * sizeof(float));
  ws.ss_bp = (float*)take(B_ * sizeof(float));
  ws.xbf   = (bf16*) take((size_t)T_ * B_ * IN_ * 2);
  ws.w_in  = (bf16*) take((size_t)X_ * IN_ * 2);
  ws.w_h   = (bf16*) take((size_t)4 * X_ * 2 * X_ * 2);
  ws.w_out = (bf16*) take((size_t)OUT_ * X_ * 2);
  ws.h0 = (bf16*)take((size_t)B_ * X_ * 2);
  ws.h1 = (bf16*)take((size_t)B_ * X_ * 2);
  ws.h2 = (bf16*)take((size_t)B_ * X_ * 2);
  ws.h3 = (bf16*)take((size_t)B_ * X_ * 2);
  ws.h4 = (bf16*)take((size_t)B_ * X_ * 2);
  ws.bp = (bf16*)take((size_t)B_ * X_ * 2);
  ws.li = (bf16*)take((size_t)B_ * X_ * 2);

  k_prep_weights<<<2048, NTHR_, 0, stream>>>(W_in, W_h, W_out, ws);
  k_prep_x<<<512, NTHR_, 0, stream>>>(x, ws);
  k_prep_state<<<512, NTHR_, 0, stream>>>(ws);
  bnn_persistent<<<NWG_, NTHR_, 0, stream>>>(ws, b_in, b_h, b_out, act_ids, out_act_ids, y);
}
